// MonteCarloSameInfluenceProcess_3410204033610
// MI455X (gfx1250) — compile-verified, matrix operand order checked by reading, not run on hardware
//
#include <hip/hip_runtime.h>
#include <hip/hip_bf16.h>

typedef __attribute__((ext_vector_type(16))) _Float16 v16h;
typedef __attribute__((ext_vector_type(8)))  float    v8f;

#define T_ENDC     100.0f
#define EPSC       1e-10f
#define NMC        15
#define BB         16
#define LL         384
#define HH         128
#define PAIRS      73536                    // 383*384/2
#define NPAIR_ROWS (BB * PAIRS)             // 1,176,576 (32-aligned)
#define NMC_ROWS   (NMC * BB * LL)          // 92,160   (32-aligned)
#define NROWS      (NPAIR_ROWS + NMC_ROWS)  // 1,268,736
#define NJOBS      (NROWS / 32)             // 39,648  (one job = 32 rows = 2 WMMA tiles)
#define WAVES_PER_BLOCK 4
#define NBLOCKS    (NJOBS / WAVES_PER_BLOCK) // 9,912

union AFrag { v16h v; _Float16 h[16]; uint4 q[2]; };
union CFrag { v8f  v; float    f[8]; };

__device__ __forceinline__ float fast_tanh(float x) {
    // tanh(x) = 1 - 2/(e^{2x}+1); e=inf -> 1, e=0 -> -1 (saturates correctly)
    float e = __expf(2.0f * x);
    return 1.0f - 2.0f * __builtin_amdgcn_rcpf(e + 1.0f);
}

__device__ __forceinline__ float softplus(float x) {
    float ax = fabsf(x);
    return fmaxf(x, 0.0f) + __logf(1.0f + __expf(-ax));
}

// invert p = r(r+1)/2 + c, c <= r  (np.tril_indices enumeration order)
__device__ __forceinline__ void invert_tri(int p, int& rr, int& cc) {
    int r = (int)floorf((sqrtf(8.0f * (float)p + 1.0f) - 1.0f) * 0.5f);
    while ((r + 1) * (r + 2) / 2 <= p) ++r;
    while (r * (r + 1) / 2 > p) --r;
    rr = r;
    cc = p - r * (r + 1) / 2;
}

__device__ __forceinline__ float mc_uniform(unsigned int m) {
    unsigned int h = m * 2654435761u + 0x9E3779B9u;
    h ^= h >> 16; h *= 0x85EBCA6Bu;
    h ^= h >> 13; h *= 0xC2B2AE35u;
    h ^= h >> 16;
    return (float)(h >> 8) * (1.0f / 16777216.0f);   // [0,1)
}

// per-row scalar MLP input
__device__ __forceinline__ float decode_x(int row, bool is_pair,
                                          const float* __restrict__ t,
                                          const int* __restrict__ lens) {
    if (is_pair) {
        int b = row / PAIRS, pp = row - b * PAIRS;
        int rr, cc; invert_tri(pp, rr, cc);
        return t[b * LL + rr + 1] - t[b * LL + cc];
    } else {
        int m   = row - NPAIR_ROWS;
        int rem = m % (BB * LL);
        int b = rem / LL, l = rem - b * LL;
        float dl = (l < lens[b]) ? (T_ENDC - t[b * LL + l]) : 0.0f;
        return mc_uniform((unsigned int)m) * dl;
    }
}

// ---------------------------------------------------------------------------
// Prep: zero accumulators; repack W2/W3 (f32, [k][n] row-major) into the exact
// per-lane WMMA B-fragment order so the MLP kernel loads a fragment as two
// consecutive b128 loads:  W?p[((kc*8+nt)*32 + lane)*16 + e]
// with k = kc*32 + ((lane&16)>>1) + e + (e&8),  n = nt*16 + (lane&15).
// ---------------------------------------------------------------------------
__global__ void prep_kernel(const float* __restrict__ W2, const float* __restrict__ W3,
                            _Float16* __restrict__ W2p, _Float16* __restrict__ W3p,
                            float* __restrict__ lambs, float* __restrict__ ints) {
    int idx = blockIdx.x * blockDim.x + threadIdx.x;
    if (idx < BB * LL) lambs[idx] = 0.0f;
    if (idx == 0) *ints = 0.0f;
    for (int i = idx; i < 2 * HH * HH; i += gridDim.x * blockDim.x) {
        int layer = i >> 14;            // 0 -> W2, 1 -> W3
        int j     = i & 16383;
        int e     = j & 15;
        int lane  = (j >> 4) & 31;
        int tnk   = j >> 9;             // kc*8 + nt
        int kc = tnk >> 3, nt = tnk & 7;
        int k = kc * 32 + ((lane & 16) >> 1) + e + (e & 8);
        int n = nt * 16 + (lane & 15);
        float w = layer ? W3[k * HH + n] : W2[k * HH + n];
        (layer ? W3p : W2p)[j] = (_Float16)w;
    }
}

// ---------------------------------------------------------------------------
// Main MLP kernel: one wave processes 32 scalar inputs (2 WMMA row-tiles)
// through the MLP; each B-fragment load feeds TWO v_wmma_f32_16x16x32_f16,
// so fragment-load latency overlaps independent WMMA work.
// ---------------------------------------------------------------------------
__global__ void __launch_bounds__(WAVES_PER_BLOCK * 32)
mlp_kernel(const float* __restrict__ t, const int* __restrict__ lens,
           const float* __restrict__ W1, const float* __restrict__ b1,
           const float* __restrict__ b2, const float* __restrict__ b3,
           const float* __restrict__ W4, const float* __restrict__ b4,
           const _Float16* __restrict__ W2p, const _Float16* __restrict__ W3p,
           float* __restrict__ lambs, float* __restrict__ ints) {
    __shared__ _Float16 stage[WAVES_PER_BLOCK][2][16 * HH];  // 32 KB transpose staging

    const int lane = threadIdx.x & 31;
    const int wv   = threadIdx.x >> 5;
    const int job  = blockIdx.x * WAVES_PER_BLOCK + wv;
    if (job >= NJOBS) return;

    const bool is_pair   = (job * 32) < NPAIR_ROWS;
    const int  ncol      = lane & 15;
    const int  koff_lane = (lane & 16) >> 1;   // 0 for lanes 0-15, 8 for 16-31
    const int  row0      = job * 32 + ncol;    // tile 0 row for this lane
    const int  row1      = row0 + 16;          // tile 1 row

    // warm the packed-weight lines (global_prefetch_b8)
    __builtin_prefetch(W2p + (lane << 4), 0, 3);
    __builtin_prefetch(W3p + (lane << 4), 0, 3);

    // ---- per-lane scalar inputs ----
    float x0 = decode_x(row0, is_pair, t, lens);
    float x1 = decode_x(row1, is_pair, t, lens);

    // ---- layer 1: build A-fragments directly (16x32 f16 layout per K-chunk) ----
    AFrag a0[4], a1[4];
#pragma unroll
    for (int kc = 0; kc < 4; ++kc) {
#pragma unroll
        for (int e = 0; e < 16; ++e) {
            int k = kc * 32 + koff_lane + e + (e & 8);
            float w = W1[k], b = b1[k];
            a0[kc].h[e] = (_Float16)fast_tanh(fmaf(x0, w, b));
            a1[kc].h[e] = (_Float16)fast_tanh(fmaf(x1, w, b));
        }
    }

    // ---- layer 2: h2 = tanh(h1 @ W2 + b2), staged to LDS for transpose ----
    const int rbase = (lane >> 4) * 8;   // D layout: lanes 0-15 rows 0-7, lanes 16-31 rows 8-15
#pragma unroll
    for (int nt = 0; nt < 8; ++nt) {
        float bias = b2[nt * 16 + ncol];
        CFrag c0, c1;
#pragma unroll
        for (int r = 0; r < 8; ++r) { c0.f[r] = bias; c1.f[r] = bias; }
#pragma unroll
        for (int kc = 0; kc < 4; ++kc) {
            AFrag bw;
            const uint4* wp = (const uint4*)(W2p + (((kc * 8 + nt) * 32 + lane) << 4));
            bw.q[0] = wp[0]; bw.q[1] = wp[1];
            c0.v = __builtin_amdgcn_wmma_f32_16x16x32_f16(
                false, a0[kc].v, false, bw.v, (short)0, c0.v, false, false);
            c1.v = __builtin_amdgcn_wmma_f32_16x16x32_f16(
                false, a1[kc].v, false, bw.v, (short)0, c1.v, false, false);
        }
#pragma unroll
        for (int r = 0; r < 8; ++r) {
            int si = (rbase + r) * HH + nt * 16 + ncol;
            stage[wv][0][si] = (_Float16)fast_tanh(c0.f[r]);
            stage[wv][1][si] = (_Float16)fast_tanh(c1.f[r]);
        }
    }

    // ---- rebuild A-fragments for layer 3 from LDS (per-wave, DS ops in-order) ----
#pragma unroll
    for (int kc = 0; kc < 4; ++kc) {
        int so = ncol * HH + kc * 32 + koff_lane;
        const _Float16* sp0 = &stage[wv][0][so];
        const _Float16* sp1 = &stage[wv][1][so];
        a0[kc].q[0] = *(const uint4*)sp0;          // k = koff..koff+7
        a0[kc].q[1] = *(const uint4*)(sp0 + 16);   // k = koff+16..koff+23
        a1[kc].q[0] = *(const uint4*)sp1;
        a1[kc].q[1] = *(const uint4*)(sp1 + 16);
    }

    // ---- layer 3 + layer 4 dot with W4 fused ----
    float p0[8], p1[8];
#pragma unroll
    for (int r = 0; r < 8; ++r) { p0[r] = 0.0f; p1[r] = 0.0f; }
#pragma unroll
    for (int nt = 0; nt < 8; ++nt) {
        float bias = b3[nt * 16 + ncol];
        CFrag c0, c1;
#pragma unroll
        for (int r = 0; r < 8; ++r) { c0.f[r] = bias; c1.f[r] = bias; }
#pragma unroll
        for (int kc = 0; kc < 4; ++kc) {
            AFrag bw;
            const uint4* wp = (const uint4*)(W3p + (((kc * 8 + nt) * 32 + lane) << 4));
            bw.q[0] = wp[0]; bw.q[1] = wp[1];
            c0.v = __builtin_amdgcn_wmma_f32_16x16x32_f16(
                false, a0[kc].v, false, bw.v, (short)0, c0.v, false, false);
            c1.v = __builtin_amdgcn_wmma_f32_16x16x32_f16(
                false, a1[kc].v, false, bw.v, (short)0, c1.v, false, false);
        }
        float w4v = W4[nt * 16 + ncol];
#pragma unroll
        for (int r = 0; r < 8; ++r) {
            p0[r] = fmaf(fast_tanh(c0.f[r]), w4v, p0[r]);
            p1[r] = fmaf(fast_tanh(c1.f[r]), w4v, p1[r]);
        }
    }

    // reduce across the 16 lanes holding columns of the same row-half
#pragma unroll
    for (int off = 8; off >= 1; off >>= 1) {
#pragma unroll
        for (int r = 0; r < 8; ++r) {
            p0[r] += __shfl_xor(p0[r], off, 32);
            p1[r] += __shfl_xor(p1[r], off, 32);
        }
    }

    // ---- emit: lanes 0 (rows 0-7) and 16 (rows 8-15) of each tile ----
    if (ncol == 0) {
        float b4v = b4[0];
        float acc = 0.0f;
#pragma unroll 1
        for (int ti = 0; ti < 2; ++ti) {
#pragma unroll 1
            for (int r = 0; r < 8; ++r) {
                int row = job * 32 + ti * 16 + rbase + r;
                float f = softplus((ti ? p1[r] : p0[r]) + b4v);
                if (is_pair) {
                    int b = row / PAIRS, pp = row - b * PAIRS;
                    int rr, cc; invert_tri(pp, rr, cc);
                    atomicAdd(&lambs[b * LL + rr + 1], f);
                } else {
                    int m   = row - NPAIR_ROWS;
                    int rem = m % (BB * LL);
                    int b = rem / LL, l = rem - b * LL;
                    float dl = (l < lens[b]) ? (T_ENDC - t[b * LL + l]) : 0.0f;
                    acc = fmaf(dl, f + EPSC, acc);
                }
            }
        }
        if (!is_pair) atomicAdd(ints, acc * (1.0f / (float)NMC));
    }
}

// ---------------------------------------------------------------------------
// Finalize: sum_log over valid entries, combine with MC integral -> nll
// ---------------------------------------------------------------------------
__global__ void finalize_kernel(const float* __restrict__ lambs, const float* __restrict__ ints,
                                const int* __restrict__ lens, const float* __restrict__ bg,
                                float* __restrict__ out) {
    __shared__ float red[256];
    float bgv = bg[0];
    float s = 0.0f;
    for (int idx = threadIdx.x; idx < BB * LL; idx += 256) {
        int b = idx / LL, l = idx - b * LL;
        if (l < lens[b]) s += logf(lambs[idx] + bgv);
    }
    red[threadIdx.x] = s;
    __syncthreads();
    for (int off = 128; off > 0; off >>= 1) {
        if (threadIdx.x < off) red[threadIdx.x] += red[threadIdx.x + off];
        __syncthreads();
    }
    if (threadIdx.x == 0) {
        float ints_total = *ints + (float)BB * T_ENDC * bgv;
        out[0] = -(red[0] - ints_total) / (float)BB;
    }
}

extern "C" void kernel_launch(void* const* d_in, const int* in_sizes, int n_in,
                              void* d_out, int out_size, void* d_ws, size_t ws_size,
                              hipStream_t stream) {
    const float* seq_pads   = (const float*)d_in[0];
    const int*   seq_lens   = (const int*)d_in[1];
    const float* background = (const float*)d_in[2];
    const float* W1 = (const float*)d_in[3];
    const float* b1 = (const float*)d_in[4];
    const float* W2 = (const float*)d_in[5];
    const float* b2 = (const float*)d_in[6];
    const float* W3 = (const float*)d_in[7];
    const float* b3 = (const float*)d_in[8];
    const float* W4 = (const float*)d_in[9];
    const float* b4 = (const float*)d_in[10];

    _Float16* W2p  = (_Float16*)d_ws;                              // 32 KB
    _Float16* W3p  = (_Float16*)((char*)d_ws + 32768);             // 32 KB
    float*    lambs = (float*)((char*)d_ws + 65536);               // 24 KB
    float*    ints  = (float*)((char*)d_ws + 65536 + 24576);       // 4 B

    prep_kernel<<<128, 256, 0, stream>>>(W2, W3, W2p, W3p, lambs, ints);
    mlp_kernel<<<NBLOCKS, WAVES_PER_BLOCK * 32, 0, stream>>>(
        seq_pads, seq_lens, W1, b1, b2, b3, W4, b4, W2p, W3p, lambs, ints);
    finalize_kernel<<<1, 256, 0, stream>>>(lambs, ints, seq_lens, background, (float*)d_out);
}
